// GroupRopeAttention_59992103191028
// MI455X (gfx1250) — matrix operand order repaired and checked by reading, not compiled, not run on hardware
//
#include <hip/hip_runtime.h>
#include <math.h>

#define EMB   2048
#define HD    128
#define NH    16
#define LSEQ  2048
#define BATCH 2
#define NQK   (NH*HD)            // 2048 Q columns
#define NTOT  (NH*HD + HD + HD)  // 2304 fused output columns
#define MROWS (BATCH*LSEQ)       // 4096 rows of x

typedef __bf16        v16bf __attribute__((ext_vector_type(16)));
typedef float         v8f   __attribute__((ext_vector_type(8)));
typedef unsigned int  u32x4 __attribute__((ext_vector_type(4)));

union BFrag { v16bf v; u32x4 q[2]; };

// float -> bf16 bits, round-to-nearest-even
__device__ inline unsigned short f2bf_bits(float f) {
  unsigned u = __builtin_bit_cast(unsigned, f);
  u += 0x7fffu + ((u >> 16) & 1u);
  return (unsigned short)(u >> 16);
}

// 32B contiguous (B-fragment: 16 consecutive bf16)
__device__ inline v16bf load_frag_b(const unsigned short* p) {
  BFrag u;
  u.q[0] = *reinterpret_cast<const u32x4*>(p);
  u.q[1] = *reinterpret_cast<const u32x4*>(p + 8);
  return u.v;
}
// A-fragment: 32B contiguous, elements {0..15} relative to p
__device__ inline v16bf load_frag_a(const unsigned short* p) {
  BFrag u;
  u.q[0] = *reinterpret_cast<const u32x4*>(p);
  u.q[1] = *reinterpret_cast<const u32x4*>(p + 8);
  return u.v;
}

__device__ inline v8f vzero8() {
  v8f z;
  #pragma unroll
  for (int i = 0; i < 8; ++i) z[i] = 0.f;
  return z;
}

__device__ inline float half16max(float v) {
  v = fmaxf(v, __shfl_xor(v, 1, 32));
  v = fmaxf(v, __shfl_xor(v, 2, 32));
  v = fmaxf(v, __shfl_xor(v, 4, 32));
  v = fmaxf(v, __shfl_xor(v, 8, 32));
  return v;
}
__device__ inline float half16sum(float v) {
  v += __shfl_xor(v, 1, 32);
  v += __shfl_xor(v, 2, 32);
  v += __shfl_xor(v, 4, 32);
  v += __shfl_xor(v, 8, 32);
  return v;
}

// ---------------- prep kernels ----------------
__global__ void k_cvt_x(const float* __restrict__ x, unsigned short* __restrict__ xb, int n) {
  int i = blockIdx.x * blockDim.x + threadIdx.x;
  if (i < n) xb[i] = f2bf_bits(x[i]);
}

__global__ void k_build_wt(const float* __restrict__ Wq, const float* __restrict__ Wk,
                           const float* __restrict__ Wv, unsigned short* __restrict__ WT) {
  int i = blockIdx.x * blockDim.x + threadIdx.x;   // over NTOT*EMB
  if (i >= NTOT * EMB) return;
  int nn = i / EMB, k = i - nn * EMB;
  float v;
  if (nn < NQK)           v = Wq[k * NQK + nn];
  else if (nn < NQK + HD) v = Wk[k * HD + (nn - NQK)];
  else                    v = Wv[k * HD + (nn - NQK - HD)];
  WT[i] = f2bf_bits(v);
}

// ---------------- fused QKV projection + RoPE ----------------
// wave computes a 16(M) x 64(N) tile; 8 waves/block; jobs = 256 Mtiles * 36 Ngroups
__global__ __launch_bounds__(256) void k_proj(const unsigned short* __restrict__ xb,
                                              const unsigned short* __restrict__ WT,
                                              unsigned short* __restrict__ Qb,
                                              unsigned short* __restrict__ Kb,
                                              unsigned short* __restrict__ Vt) {
  const int lane = threadIdx.x & 31;
  const int wave = threadIdx.x >> 5;
  const int job  = blockIdx.x * 8 + wave;
  const int mt   = job / 36;
  const int ng   = job - mt * 36;
  const int hl   = lane >> 4;
  const int ln   = lane & 15;

  v8f acc[4];
  #pragma unroll
  for (int t = 0; t < 4; ++t) acc[t] = vzero8();

  const unsigned short* arow = xb + (size_t)(mt * 16 + ln) * EMB;
  for (int kk = 0; kk < EMB; kk += 32) {
    v16bf afrag = load_frag_a(arow + kk + hl * 16);
    #pragma unroll
    for (int t = 0; t < 4; ++t) {
      int n = ng * 64 + t * 16 + ln;
      v16bf bfrag = load_frag_b(WT + (size_t)n * EMB + kk + hl * 16);
      acc[t] = __builtin_amdgcn_wmma_f32_16x16x32_bf16(
          false, afrag, false, bfrag, (short)0, acc[t], false, false);
    }
  }

  const float qscale = 0.08838834764831845f;  // 1/sqrt(HD)
  const float lnth   = 0.14391156831133613f;  // ln(10000)/64
  #pragma unroll
  for (int t = 0; t < 4; ++t) {
    int n = ng * 64 + t * 16 + ln;   // branch below is wave-uniform (16-col groups)
    #pragma unroll
    for (int r = 0; r < 8; ++r) {
      int m = mt * 16 + r + hl * 8;
      int b = m >> 11;
      int l = m & (LSEQ - 1);
      float v = acc[t][r];
      float partner = __shfl_xor(v, 1, 32);  // value at column n^1 (adjacent lane)
      if (n < NQK) {
        int d = n & (HD - 1);
        float inv = __expf(-(float)((n >> 1) & 63) * lnth);
        float s, c; sincosf((float)l * inv, &s, &c);
        float rot = (n & 1) ? partner : -partner;
        int h = (n >> 7) & (NH - 1);
        Qb[(((size_t)(b * NH + h) * LSEQ + l) << 7) + d] = f2bf_bits((v * c + rot * s) * qscale);
      } else if (n < NQK + HD) {
        int d = n - NQK;
        float inv = __expf(-(float)((d >> 1) & 63) * lnth);
        float s, c; sincosf((float)l * inv, &s, &c);
        float rot = (d & 1) ? partner : -partner;
        Kb[(((size_t)b * LSEQ + l) << 7) + d] = f2bf_bits(v * c + rot * s);
      } else {
        int d = n - NQK - HD;
        Vt[(size_t)(b * HD + d) * LSEQ + l] = f2bf_bits(v);  // transposed store
      }
    }
  }
}

// ---------------- flash attention ----------------
// one wave per 16-query tile; 8 waves/block; grid = B*NH*16
__global__ __launch_bounds__(256) void k_attn(const unsigned short* __restrict__ Qb,
                                              const unsigned short* __restrict__ Kb,
                                              const unsigned short* __restrict__ Vt,
                                              float* __restrict__ out) {
  __shared__ __align__(32) unsigned short ldsP[8][16][32];  // per-wave P-transpose tile
  const int lane = threadIdx.x & 31;
  const int wave = threadIdx.x >> 5;
  const int hl   = lane >> 4;
  const int ln   = lane & 15;
  const int bh   = blockIdx.x >> 4;      // 0..31
  const int qg   = blockIdx.x & 15;
  const int b    = bh >> 4;
  const int h    = bh & (NH - 1);
  const int qt   = qg * 8 + wave;        // 0..127

  // Q A-fragments (4 d-chunks of 32), Q already scaled by 1/sqrt(HD)
  const unsigned short* qrow = Qb + (((size_t)(b * NH + h) * LSEQ + qt * 16 + ln) << 7);
  v16bf qf[4];
  #pragma unroll
  for (int c = 0; c < 4; ++c) qf[c] = load_frag_a(qrow + c * 32 + hl * 16);

  v8f accv[8];
  float mrow[8], lrow[8];
  #pragma unroll
  for (int c = 0; c < 8; ++c) accv[c] = vzero8();
  #pragma unroll
  for (int r = 0; r < 8; ++r) { mrow[r] = -__builtin_inff(); lrow[r] = 0.f; }

  const unsigned short* kbase = Kb + ((size_t)(b * LSEQ) << 7);
  const unsigned short* vbase = Vt + (size_t)(b * HD) * LSEQ;
  const int nkt = ((qt * 16 + 15) >> 5) + 1;   // causal bound, 32-key tiles

  for (int kt = 0; kt < nkt; ++kt) {
    const int kt0 = kt << 5;
    v8f S0 = vzero8(), S1 = vzero8();
    #pragma unroll
    for (int c = 0; c < 4; ++c) {
      v16bf kf0 = load_frag_b(kbase + ((size_t)(kt0 + ln) << 7) + c * 32 + hl * 16);
      S0 = __builtin_amdgcn_wmma_f32_16x16x32_bf16(false, qf[c], false, kf0, (short)0, S0, false, false);
      v16bf kf1 = load_frag_b(kbase + ((size_t)(kt0 + 16 + ln) << 7) + c * 32 + hl * 16);
      S1 = __builtin_amdgcn_wmma_f32_16x16x32_bf16(false, qf[c], false, kf1, (short)0, S1, false, false);
    }
    if (kt + 1 < nkt)   // pull next K tile toward the WGP
      __builtin_prefetch(kbase + ((size_t)(kt0 + 32 + ln) << 7), 0, 1);

    // causal mask + online softmax (per-row state lives replicated in each 16-lane half)
    const int qg0 = qt * 16 + hl * 8;
    #pragma unroll
    for (int r = 0; r < 8; ++r) {
      int qq = qg0 + r;
      float s0 = (kt0 + ln      > qq) ? -__builtin_inff() : S0[r];
      float s1 = (kt0 + 16 + ln > qq) ? -__builtin_inff() : S1[r];
      float mx = half16max(fmaxf(s0, s1));
      float mnew = fmaxf(mrow[r], mx);
      float sc = __expf(mrow[r] - mnew);
      float p0 = __expf(s0 - mnew);
      float p1 = __expf(s1 - mnew);
      float rs = half16sum(p0 + p1);
      lrow[r] = lrow[r] * sc + rs;
      mrow[r] = mnew;
      #pragma unroll
      for (int c = 0; c < 8; ++c) accv[c][r] *= sc;
      ldsP[wave][r + hl * 8][ln]      = f2bf_bits(p0);
      ldsP[wave][r + hl * 8][ln + 16] = f2bf_bits(p1);
    }
    // same-wave DS ops are in-order (ISA §7.3); just block compiler reordering
    asm volatile("" ::: "memory");
    v16bf pf = load_frag_a(&ldsP[wave][ln][hl * 16]);

    #pragma unroll
    for (int c = 0; c < 8; ++c) {
      v16bf vf = load_frag_b(vbase + (size_t)(c * 16 + ln) * LSEQ + kt0 + hl * 16);
      accv[c] = __builtin_amdgcn_wmma_f32_16x16x32_bf16(false, pf, false, vf, (short)0, accv[c], false, false);
    }
  }

  #pragma unroll
  for (int r = 0; r < 8; ++r) {
    int q = qt * 16 + r + hl * 8;
    float invl = 1.f / lrow[r];
    float* orow = out + (size_t)(b * LSEQ + q) * (NH * HD) + h * HD;
    #pragma unroll
    for (int c = 0; c < 8; ++c)
      orow[c * 16 + ln] = accv[c][r] * invl;
  }
}

extern "C" void kernel_launch(void* const* d_in, const int* in_sizes, int n_in,
                              void* d_out, int out_size, void* d_ws, size_t ws_size,
                              hipStream_t stream) {
  (void)in_sizes; (void)n_in; (void)out_size; (void)ws_size;
  const float* x  = (const float*)d_in[0];
  const float* Wq = (const float*)d_in[1];
  const float* Wk = (const float*)d_in[2];
  const float* Wv = (const float*)d_in[3];
  float* out = (float*)d_out;

  unsigned short* xb = (unsigned short*)d_ws;                  // 4096*2048
  unsigned short* WT = xb + (size_t)MROWS * EMB;               // 2304*2048
  unsigned short* Qb = WT + (size_t)NTOT * EMB;                // B*NH*L*HD (rope'd, pre-scaled)
  unsigned short* Kb = Qb + (size_t)BATCH * NH * LSEQ * HD;    // B*L*HD (rope'd)
  unsigned short* Vt = Kb + (size_t)BATCH * LSEQ * HD;         // B*HD*L (transposed)

  k_cvt_x<<<(MROWS * EMB) / 256, 256, 0, stream>>>(x, xb, MROWS * EMB);
  k_build_wt<<<(NTOT * EMB) / 256, 256, 0, stream>>>(Wq, Wk, Wv, WT);
  k_proj<<<(256 * 36) / 8, 256, 0, stream>>>(xb, WT, Qb, Kb, Vt);
  k_attn<<<BATCH * NH * 16, 256, 0, stream>>>(Qb, Kb, Vt, out);
}
